// PostProcessor_8091718386062
// MI455X (gfx1250) — compile-verified, matrix operand order checked by reading, not run on hardware
//
#include <hip/hip_runtime.h>
#include <math.h>

typedef __attribute__((ext_vector_type(16))) _Float16 v16h;
typedef __attribute__((ext_vector_type(8)))  float    v8f;

#define RDIM   1000
#define CDIM   81
#define NCLS   80
#define KDET   100
#define SCORE_TH 0.05f
#define NMS_TH   0.5f
#define NEGV    (-1e9f)
#define XFORM_CLIP 4.135166556742356f   /* log(1000/16) */

// ---------------------------------------------------------------------------
// Kernel 1: softmax over 81 classes; row-sum of exp() on the WMMA pipe.
// One wave per 16-row tile. A = exp tile (16x32 f16, Dekker hi/lo split),
// B = ones(32x16) f16, C/D = f32 accumulator -> D[m,n] = rowsum(m).
// Padded rows are clamped to row RDIM-1 and zeroed via rowscale so all
// global loads are unconditional (wide b128 loads, no exec-mask ladders).
// ---------------------------------------------------------------------------
__global__ void __launch_bounds__(32)
softmax_wmma_kernel(const float* __restrict__ logits, float* __restrict__ scores_t) {
    const int lane    = threadIdx.x;          // 0..31
    const int rowbase = blockIdx.x * 16;
    const int myrow   = rowbase + (lane & 15);
    const int crow    = (myrow < RDIM) ? myrow : (RDIM - 1);
    const float rowscale = (myrow < RDIM) ? 1.0f : 0.0f;
    const float* __restrict__ lp = logits + crow * CDIM;

    // --- row max: every lane computes it for its (clamped) row; lanes L and
    //     L+16 redundantly compute the same value -> uniform flow, no shfl ---
    float rmax = lp[0];
    for (int c = 1; c < CDIM; ++c) rmax = fmaxf(rmax, lp[c]);

    // --- B = ones(32x16) f16 (layout-invariant: all elements equal) ---
    v16h bones;
#pragma unroll
    for (int j = 0; j < 16; ++j) bones[j] = (_Float16)1.0f;

    // 16-bit A 16x32 layout (ISA 7.12.2): lane L holds row M = L&15;
    // halves j map to K = kbase + (j<8 ? j : j+8), kbase = (lane<16 ? 0 : 8).
    // Valid-column check is compile-time except for the kbase offset, which we
    // fold in: k_lo = j (+32 per chunk), columns >= CDIM-kbase are zero.
    v8f acc = {};
    const int kbase = (lane < 16) ? 0 : 8;
#pragma unroll
    for (int chunk = 0; chunk < 3; ++chunk) {
        v16h ahi, alo;
#pragma unroll
        for (int j = 0; j < 16; ++j) {
            const int krel = chunk * 32 + ((j < 8) ? j : (j + 8)); // 0..95, lane-invariant part
            float v = 0.0f;
            if (krel < CDIM) {                 // worst case (kbase=8) still in-bounds?
                const int k = krel + kbase;    // actual column
                // krel < 81 guarantees k <= 88; clamp check on k must also hold.
                // Columns with k >= CDIM contribute zero; this test is
                // uniform per (j,chunk) within each 16-lane half.
                const float x = (k < CDIM) ? lp[k] : 0.0f;
                v = (k < CDIM) ? __expf(x - rmax) * rowscale : 0.0f;
            }
            const _Float16 hi = (_Float16)v;
            ahi[j] = hi;
            alo[j] = (_Float16)(v - (float)hi);   // residual -> ~f32 accuracy
        }
        acc = __builtin_amdgcn_wmma_f32_16x16x32_f16(false, ahi, false, bones,
                                                     (short)0, acc, false, false);
        acc = __builtin_amdgcn_wmma_f32_16x16x32_f16(false, alo, false, bones,
                                                     (short)0, acc, false, false);
    }

    // --- extract row sums: lane0 holds D[M=0..7,N=0], lane16 holds D[M=8..15,N=0]
    __shared__ float sums[16];
    if (lane == 0) {
#pragma unroll
        for (int m = 0; m < 8; ++m) sums[m] = acc[m];
    }
    if (lane == 16) {
#pragma unroll
        for (int m = 0; m < 8; ++m) sums[8 + m] = acc[m];
    }
    __syncthreads();

    // --- write probs for classes 1..80, transposed [class][row] ---
    if (lane < 16 && myrow < RDIM) {
        const float inv = 1.0f / sums[lane];
        for (int c = 1; c < CDIM; ++c) {
            scores_t[(c - 1) * RDIM + myrow] = __expf(lp[c] - rmax) * inv;
        }
    }
}

// ---------------------------------------------------------------------------
// Kernel 2: BoxCoder.decode + clip for classes 1..80, transposed layout
// boxes_t[((c-1)*R + r)*4 + {0..3}].
// ---------------------------------------------------------------------------
__global__ void __launch_bounds__(256)
decode_kernel(const float* __restrict__ rel, const float* __restrict__ props,
              float* __restrict__ boxes_t,
              const int* __restrict__ pW, const int* __restrict__ pH) {
    const int idx = blockIdx.x * blockDim.x + threadIdx.x;
    if (idx >= NCLS * RDIM) return;
    const int c = idx / RDIM + 1;   // classes 1..80 (background dropped)
    const int r = idx - (c - 1) * RDIM;

    const float Wm1 = (float)(*pW) - 1.0f;
    const float Hm1 = (float)(*pH) - 1.0f;

    const float x1 = props[r * 4 + 0], y1 = props[r * 4 + 1];
    const float x2 = props[r * 4 + 2], y2 = props[r * 4 + 3];
    const float w  = x2 - x1 + 1.0f,  h  = y2 - y1 + 1.0f;
    const float cx = x1 + 0.5f * w,   cy = y1 + 0.5f * h;

    const float* rr = rel + r * (CDIM * 4) + c * 4;
    const float dx = rr[0] * 0.1f;
    const float dy = rr[1] * 0.1f;
    const float dw = fminf(rr[2] * 0.2f, XFORM_CLIP);
    const float dh = fminf(rr[3] * 0.2f, XFORM_CLIP);

    const float pcx = dx * w + cx, pcy = dy * h + cy;
    const float pw  = __expf(dw) * w, ph = __expf(dh) * h;

    float bx1 = pcx - 0.5f * pw, by1 = pcy - 0.5f * ph;
    float bx2 = pcx + 0.5f * pw - 1.0f, by2 = pcy + 0.5f * ph - 1.0f;
    bx1 = fminf(fmaxf(bx1, 0.0f), Wm1);
    by1 = fminf(fmaxf(by1, 0.0f), Hm1);
    bx2 = fminf(fmaxf(bx2, 0.0f), Wm1);
    by2 = fminf(fmaxf(by2, 0.0f), Hm1);

    float* ob = boxes_t + idx * 4;
    ob[0] = bx1; ob[1] = by1; ob[2] = bx2; ob[3] = by2;
}

// ---------------------------------------------------------------------------
// Kernel 3: greedy NMS per class via iterative argmax-select + suppress
// (exactly equivalent to sorted greedy NMS; ties -> lowest original index,
// matching JAX stable argsort). One block per class.
// ---------------------------------------------------------------------------
__global__ void __launch_bounds__(256)
nms_kernel(const float* __restrict__ scores_t, const float* __restrict__ boxes_t,
           float* __restrict__ flat) {
    const int cls = blockIdx.x;
    const int t   = threadIdx.x;
    const float* sc = scores_t + cls * RDIM;
    const float* bx = boxes_t + cls * RDIM * 4;

    __shared__ float         s_score[RDIM];
    __shared__ float4        s_box[RDIM];
    __shared__ unsigned char s_alive[RDIM];
    __shared__ unsigned char s_keep[RDIM];
    __shared__ float         red_v[256];
    __shared__ int           red_i[256];

    for (int i = t; i < RDIM; i += 256) {
        if (i + 256 < RDIM) __builtin_prefetch(bx + (i + 256) * 4, 0, 1);
        const float s = sc[i];
        s_score[i] = s;
        s_alive[i] = (s > SCORE_TH) ? 1 : 0;
        s_keep[i]  = 0;
        s_box[i] = make_float4(bx[i * 4 + 0], bx[i * 4 + 1],
                               bx[i * 4 + 2], bx[i * 4 + 3]);
    }
    __syncthreads();

    for (int iter = 0; iter < RDIM; ++iter) {
        // --- block-wide argmax over alive (ties -> lowest index) ---
        float bv = -3.0e38f; int bi = RDIM;
        for (int i = t; i < RDIM; i += 256) {
            if (s_alive[i]) {
                const float s = s_score[i];
                if (s > bv || (s == bv && i < bi)) { bv = s; bi = i; }
            }
        }
        red_v[t] = bv; red_i[t] = bi;
        __syncthreads();
        for (int off = 128; off > 0; off >>= 1) {
            if (t < off) {
                const float ov = red_v[t + off]; const int oi = red_i[t + off];
                if (ov > red_v[t] || (ov == red_v[t] && oi < red_i[t])) {
                    red_v[t] = ov; red_i[t] = oi;
                }
            }
            __syncthreads();
        }
        if (red_v[0] < 0.0f) break;            // no alive box left (alive => >0.05)
        const int best = red_i[0];
        const float4 bb = s_box[best];
        if (t == 0) { s_keep[best] = 1; s_alive[best] = 0; }
        __syncthreads();

        // --- suppress boxes overlapping the kept one (legacy +1 IoU) ---
        const float barea = (bb.z - bb.x + 1.0f) * (bb.w - bb.y + 1.0f);
        for (int i = t; i < RDIM; i += 256) {
            if (s_alive[i]) {
                const float4 b = s_box[i];
                const float ltx = fmaxf(bb.x, b.x), lty = fmaxf(bb.y, b.y);
                const float rbx = fminf(bb.z, b.z), rby = fminf(bb.w, b.w);
                const float iw = fmaxf(rbx - ltx + 1.0f, 0.0f);
                const float ih = fmaxf(rby - lty + 1.0f, 0.0f);
                const float inter = iw * ih;
                const float area  = (b.z - b.x + 1.0f) * (b.w - b.y + 1.0f);
                const float iou   = inter / (barea + area - inter);
                if (iou > NMS_TH) s_alive[i] = 0;
            }
        }
        __syncthreads();
    }

    for (int i = t; i < RDIM; i += 256)
        flat[cls * RDIM + i] = s_keep[i] ? s_score[i] : NEGV;
}

// ---------------------------------------------------------------------------
// Kernel 4: top-100 over flat[80000] (ties -> ascending index, matching
// lax.top_k). Single block; "taken" mask kept in LDS so global state is
// untouched (graph-replay safe). Emits scores / boxes / labels.
// ---------------------------------------------------------------------------
__global__ void __launch_bounds__(1024)
topk_kernel(const float* __restrict__ flat, const float* __restrict__ boxes_t,
            float* __restrict__ out) {
    const int N = NCLS * RDIM;
    const int t = threadIdx.x;
    __shared__ float    rv[1024];
    __shared__ int      ri[1024];
    __shared__ unsigned taken[(NCLS * RDIM + 31) / 32];

    for (int i = t; i < (N + 31) / 32; i += 1024) taken[i] = 0u;
    __syncthreads();

    for (int k = 0; k < KDET; ++k) {
        float bv = -3.0e38f; int bi = N;
        for (int i = t; i < N; i += 1024) {
            if (!(taken[i >> 5] & (1u << (i & 31)))) {
                const float s = flat[i];
                if (s > bv || (s == bv && i < bi)) { bv = s; bi = i; }
            }
        }
        rv[t] = bv; ri[t] = bi;
        __syncthreads();
        for (int off = 512; off > 0; off >>= 1) {
            if (t < off) {
                const float ov = rv[t + off]; const int oi = ri[t + off];
                if (ov > rv[t] || (ov == rv[t] && oi < ri[t])) { rv[t] = ov; ri[t] = oi; }
            }
            __syncthreads();
        }
        const int best = ri[0];
        if (t == 0) {
            out[k] = rv[0];                                   // top_s
            out[KDET + k * 4 + 0] = boxes_t[best * 4 + 0];    // top_b
            out[KDET + k * 4 + 1] = boxes_t[best * 4 + 1];
            out[KDET + k * 4 + 2] = boxes_t[best * 4 + 2];
            out[KDET + k * 4 + 3] = boxes_t[best * 4 + 3];
            out[KDET + KDET * 4 + k] = (float)(best / RDIM + 1);  // labels
            taken[best >> 5] |= (1u << (best & 31));
        }
        __syncthreads();
    }
}

// ---------------------------------------------------------------------------
extern "C" void kernel_launch(void* const* d_in, const int* in_sizes, int n_in,
                              void* d_out, int out_size, void* d_ws, size_t ws_size,
                              hipStream_t stream) {
    const float* logits = (const float*)d_in[0];   // [1000, 81]
    const float* rel    = (const float*)d_in[1];   // [1000, 324]
    const float* props  = (const float*)d_in[2];   // [1000, 4]
    const int*   pW     = (const int*)d_in[3];     // scalar
    const int*   pH     = (const int*)d_in[4];     // scalar
    float* out = (float*)d_out;                    // 100 + 400 + 100 = 600 floats

    char* ws = (char*)d_ws;
    float* scores_t = (float*)(ws);                          // 80*1000 f32
    float* boxes_t  = (float*)(ws + 320000);                 // 80*1000*4 f32
    float* flat     = (float*)(ws + 320000 + 1280000);       // 80*1000 f32

    (void)in_sizes; (void)n_in; (void)out_size; (void)ws_size;

    const int ntiles = (RDIM + 15) / 16;                     // 63 wave-tiles
    softmax_wmma_kernel<<<ntiles, 32, 0, stream>>>(logits, scores_t);
    decode_kernel<<<(NCLS * RDIM + 255) / 256, 256, 0, stream>>>(rel, props, boxes_t, pW, pH);
    nms_kernel<<<NCLS, 256, 0, stream>>>(scores_t, boxes_t, flat);
    topk_kernel<<<1, 1024, 0, stream>>>(flat, boxes_t, out);
}
